// GroupQueryAttention_6846177870521
// MI455X (gfx1250) — matrix operand order repaired and checked by reading, not compiled, not run on hardware
//
#include <hip/hip_runtime.h>
#include <hip/hip_bf16.h>

typedef __bf16 bf16;
typedef __attribute__((ext_vector_type(16))) __bf16 v16bf;
typedef __attribute__((ext_vector_type(8)))  __bf16 v8bf;
typedef __attribute__((ext_vector_type(8)))  float   v8f;

#define HIDDEN  1024
#define NHEAD   16
#define NKV     4
#define HEADDIM 64
#define KVDIM   256      // NKV * HEADDIM
#define SEQ     2048
#define BATCH   2
#define MTOT    (BATCH * SEQ)   // 4096

__device__ __forceinline__ v16bf make_frag(v8bf lo, v8bf hi) {
  v16bf r;
#pragma unroll
  for (int i = 0; i < 8; ++i) { r[i] = lo[i]; r[i + 8] = hi[i]; }
  return r;
}

__device__ __forceinline__ v8f wmma_bf16(v16bf a, v16bf b, v8f c) {
  // D = A(16x32 bf16) * B(32x16 bf16) + C(16x16 f32)
  return __builtin_amdgcn_wmma_f32_16x16x32_bf16(false, a, false, b,
                                                 (short)0, c, false, false);
}

// ---------------------------------------------------------------------------
// DPP16 row_ror reductions within each 16-lane half (wave32: 2 DPP rows).
// Replaces ds_bpermute-based shuffles: pure VALU, co-executes with XDL WMMA.
// ---------------------------------------------------------------------------
template <int CTRL>
__device__ __forceinline__ float dpp_mov_f32(float x) {
  int v = __builtin_amdgcn_update_dpp(0, __builtin_bit_cast(int, x),
                                      CTRL, 0xf, 0xf, true);
  return __builtin_bit_cast(float, v);
}

__device__ __forceinline__ float rowmax16(float x) {
  x = fmaxf(x, dpp_mov_f32<0x121>(x));   // row_ror:1
  x = fmaxf(x, dpp_mov_f32<0x122>(x));   // row_ror:2
  x = fmaxf(x, dpp_mov_f32<0x124>(x));   // row_ror:4
  x = fmaxf(x, dpp_mov_f32<0x128>(x));   // row_ror:8
  return x;
}

__device__ __forceinline__ float rowsum16(float x) {
  x += dpp_mov_f32<0x121>(x);
  x += dpp_mov_f32<0x122>(x);
  x += dpp_mov_f32<0x124>(x);
  x += dpp_mov_f32<0x128>(x);
  return x;
}

// ---------------------------------------------------------------------------
// fp32 -> bf16 conversion (grid-stride)
// ---------------------------------------------------------------------------
__global__ void cvt_f32_bf16(const float* __restrict__ s, bf16* __restrict__ d, int n) {
  int i = blockIdx.x * blockDim.x + threadIdx.x;
  int stride = gridDim.x * blockDim.x;
  for (; i < n; i += stride) d[i] = (bf16)s[i];
}

// ---------------------------------------------------------------------------
// Tiled WMMA GEMM: C[MTOT,N] = A[MTOT,1024](bf16) @ W[1024,N](bf16) + bias(f32)
// block = 128 threads (4 waves), tile 64x64, K stepped by 32.
// W chunk staged transposed in LDS; A fragments loaded directly from global.
// ---------------------------------------------------------------------------
template <bool OUT_F32>
__global__ void gemm_bias(const bf16* __restrict__ A, const bf16* __restrict__ W,
                          const float* __restrict__ bias, void* __restrict__ Cout,
                          int N) {
  __shared__ __align__(16) bf16 Wt[64][40];   // [n - n0][k - kc], padded stride

  const int lane = threadIdx.x & 31;
  const int wave = threadIdx.x >> 5;
  const int hrow = lane & 15;       // M (A) or N (B/C) index within tile
  const int hi   = lane >> 4;       // half-wave selector
  const int r0   = blockIdx.x * 64 + wave * 16;
  const int n0   = blockIdx.y * 64;

  v8f acc[4] = {};
  const size_t arow = (size_t)(r0 + hrow);

  for (int kc = 0; kc < HIDDEN; kc += 32) {
    // stage W chunk [32 x 64] transposed into LDS
    {
      const int kk = threadIdx.x & 31;
      const int nb = (threadIdx.x >> 5) * 16;
      const bf16* wp = W + (size_t)(kc + kk) * N + n0 + nb;
      v8bf w0 = *(const v8bf*)wp;
      v8bf w1 = *(const v8bf*)(wp + 8);
#pragma unroll
      for (int j = 0; j < 8; ++j) {
        Wt[nb + j][kk]     = w0[j];
        Wt[nb + 8 + j][kk] = w1[j];
      }
    }
    __syncthreads();

    // A fragment: lane holds row=hrow, K = kb..kb+7 and kb+16..kb+23
    const bf16* ap = A + arow * HIDDEN + kc + hi * 8;
    v16bf afrag = make_frag(*(const v8bf*)ap, *(const v8bf*)(ap + 16));

#pragma unroll
    for (int t = 0; t < 4; ++t) {
      // B fragment: lane holds col=hrow of subtile t, K = hi*8..hi*8+7 and hi*8+16..hi*8+23
      const bf16* bp = &Wt[t * 16 + hrow][hi * 8];
      v16bf bfrag = make_frag(*(const v8bf*)bp, *(const v8bf*)(bp + 16));
      acc[t] = wmma_bf16(afrag, bfrag, acc[t]);
    }
    __syncthreads();
  }

  // epilogue: C-tile layout row = hi*8+i, col = hrow
#pragma unroll
  for (int t = 0; t < 4; ++t) {
    const int col = n0 + t * 16 + hrow;
    const float bb = bias[col];
#pragma unroll
    for (int i = 0; i < 8; ++i) {
      const size_t row = (size_t)(r0 + hi * 8 + i);
      const float v = acc[t][i] + bb;
      if (OUT_F32) ((float*)Cout)[row * N + col] = v;
      else         ((bf16*)Cout)[row * N + col]  = (bf16)v;
    }
  }
}

// ---------------------------------------------------------------------------
// Flash GQA attention.
// grid = (SEQ/64, BATCH*NHEAD); block = 128 threads (4 waves).
// Each wave owns 16 query rows; block streams K/V in 32-key chunks via LDS.
// K chunk staged with GLOBAL_LOAD_ASYNC_TO_LDS_B128 (ASYNCcnt path); V chunk
// staged manually because it is transposed in flight.
// Q: [MTOT, HIDDEN] bf16 (col = h*64+d); K/V: [MTOT, KVDIM] bf16.
// O: [MTOT, HIDDEN] bf16.
// ---------------------------------------------------------------------------
__global__ void gqa_attn(const bf16* __restrict__ Q, const bf16* __restrict__ K,
                         const bf16* __restrict__ V, bf16* __restrict__ O) {
  __shared__ __align__(16) bf16 Kt[32][72];       // [key][d], padded
  __shared__ __align__(16) bf16 Vt[64][40];       // [d][key], transposed, padded
  __shared__ __align__(16) bf16 Pb[4][16][32];    // per-wave P relayout buffer

  const int lane = threadIdx.x & 31;
  const int wave = threadIdx.x >> 5;
  const int hrow = lane & 15;
  const int hi   = lane >> 4;

  const int bh  = blockIdx.y;            // 0..31
  const int b   = bh >> 4;
  const int h   = bh & 15;
  const int hkv = h >> 2;

  const int    q0       = blockIdx.x * 64 + wave * 16;
  const size_t qrowbase = (size_t)b * SEQ + q0;
  const size_t kvrow0   = (size_t)b * SEQ;

  v8f acc[4] = {};
  float m[8], l[8];
#pragma unroll
  for (int i = 0; i < 8; ++i) { m[i] = -3.0e38f; l[i] = 0.0f; }

  // Q fragments for d-chunks [0,32) and [32,64) — invariant over key loop
  const bf16* qp = Q + (qrowbase + hrow) * HIDDEN + h * HEADDIM + hi * 8;
  const v16bf qa0 = make_frag(*(const v8bf*)qp,        *(const v8bf*)(qp + 16));
  const v16bf qa1 = make_frag(*(const v8bf*)(qp + 32), *(const v8bf*)(qp + 48));

  // per-thread staging coordinates
  const int skey = threadIdx.x & 31;
  const int sdb  = (threadIdx.x >> 5) * 16;
  const unsigned kt_lds = (unsigned)(size_t)&Kt[skey][sdb];

  for (int kb0 = 0; kb0 < SEQ; kb0 += 32) {
    // ---- stage K chunk [32 x 64] via async global->LDS (row-major copy) ----
    {
      const size_t row = (kvrow0 + kb0 + skey) * KVDIM + hkv * HEADDIM + sdb;
      const bf16* kp = K + row;
      asm volatile("global_load_async_to_lds_b128 %0, %1, off"
                   :: "v"(kt_lds), "v"(kp) : "memory");
      asm volatile("global_load_async_to_lds_b128 %0, %1, off offset:16"
                   :: "v"(kt_lds), "v"(kp) : "memory");

      // ---- stage V chunk transposed [64 x 32] through VGPRs ----
      const bf16* vp = V + row;
      v8bf v0 = *(const v8bf*)vp;
      v8bf v1 = *(const v8bf*)(vp + 8);
#pragma unroll
      for (int j = 0; j < 8; ++j) {
        Vt[sdb + j][skey]     = v0[j];
        Vt[sdb + 8 + j][skey] = v1[j];
      }
      if (kb0 + 32 < SEQ) {  // prefetch next chunk
        __builtin_prefetch(kp + 32 * KVDIM, 0, 0);
        __builtin_prefetch(vp + 32 * KVDIM, 0, 0);
      }
      asm volatile("s_wait_asynccnt 0x0" ::: "memory");
    }
    __syncthreads();

    // ---- scores: S = Q (16x64) * K^T (64x32), two 16x16 tiles ----
    v8f s[2];
#pragma unroll
    for (int t = 0; t < 2; ++t) {
      v8f sc = {};
      {
        const bf16* bp = &Kt[t * 16 + hrow][hi * 8];          // d-chunk 0
        sc = wmma_bf16(qa0, make_frag(*(const v8bf*)bp, *(const v8bf*)(bp + 16)), sc);
      }
      {
        const bf16* bp = &Kt[t * 16 + hrow][32 + hi * 8];     // d-chunk 1
        sc = wmma_bf16(qa1, make_frag(*(const v8bf*)bp, *(const v8bf*)(bp + 16)), sc);
      }
      s[t] = sc;
    }

    // ---- online softmax over 32 keys (rows live in 16-lane halves) ----
    const float scale = 0.125f;   // 1/sqrt(64)
#pragma unroll
    for (int i = 0; i < 8; ++i) {
      float a0 = s[0][i] * scale;
      float a1 = s[1][i] * scale;
      const float mx = rowmax16(fmaxf(a0, a1));
      const float mn   = fmaxf(m[i], mx);
      const float corr = __expf(m[i] - mn);
      const float p0   = __expf(a0 - mn);
      const float p1   = __expf(a1 - mn);
      const float rs   = rowsum16(p0 + p1);
      l[i] = l[i] * corr + rs;
      m[i] = mn;
#pragma unroll
      for (int t = 0; t < 4; ++t) acc[t][i] *= corr;
      const int row = hi * 8 + i;
      Pb[wave][row][hrow]      = (bf16)p0;
      Pb[wave][row][16 + hrow] = (bf16)p1;
    }

    // ---- acc += P (16x32) * V (32x64); same-wave LDS ops are in-order ----
    const bf16* pp = &Pb[wave][hrow][hi * 8];
    const v16bf pfrag = make_frag(*(const v8bf*)pp, *(const v8bf*)(pp + 16));
#pragma unroll
    for (int t = 0; t < 4; ++t) {
      const bf16* bp = &Vt[t * 16 + hrow][hi * 8];
      acc[t] = wmma_bf16(pfrag, make_frag(*(const v8bf*)bp, *(const v8bf*)(bp + 16)), acc[t]);
    }
    __syncthreads();
  }

  // ---- normalize and write O (bf16) ----
#pragma unroll
  for (int t = 0; t < 4; ++t) {
#pragma unroll
    for (int i = 0; i < 8; ++i) {
      const int row = hi * 8 + i;
      const float v = acc[t][i] / l[i];
      O[(qrowbase + row) * HIDDEN + h * HEADDIM + t * 16 + hrow] = (bf16)v;
    }
  }
}

// ---------------------------------------------------------------------------
// Host launcher
// ---------------------------------------------------------------------------
extern "C" void kernel_launch(void* const* d_in, const int* in_sizes, int n_in,
                              void* d_out, int out_size, void* d_ws, size_t ws_size,
                              hipStream_t stream) {
  (void)in_sizes; (void)n_in; (void)out_size; (void)ws_size;

  const float* X  = (const float*)d_in[0];
  const float* Wq = (const float*)d_in[1];
  const float* bq = (const float*)d_in[2];
  const float* Wk = (const float*)d_in[3];
  const float* bk = (const float*)d_in[4];
  const float* Wv = (const float*)d_in[5];
  const float* bv = (const float*)d_in[6];
  const float* Wo = (const float*)d_in[7];
  const float* bo = (const float*)d_in[8];
  float* out = (float*)d_out;

  // workspace carve-up (bf16 buffers), ~34.6 MB total
  char* p = (char*)d_ws;
  bf16* Xb  = (bf16*)p; p += (size_t)MTOT * HIDDEN * 2;
  bf16* Qb  = (bf16*)p; p += (size_t)MTOT * HIDDEN * 2;
  bf16* Ob  = (bf16*)p; p += (size_t)MTOT * HIDDEN * 2;
  bf16* Kb  = (bf16*)p; p += (size_t)MTOT * KVDIM * 2;
  bf16* Vb  = (bf16*)p; p += (size_t)MTOT * KVDIM * 2;
  bf16* Wqb = (bf16*)p; p += (size_t)HIDDEN * HIDDEN * 2;
  bf16* Wob = (bf16*)p; p += (size_t)HIDDEN * HIDDEN * 2;
  bf16* Wkb = (bf16*)p; p += (size_t)HIDDEN * KVDIM * 2;
  bf16* Wvb = (bf16*)p; p += (size_t)HIDDEN * KVDIM * 2;

  const dim3 cblk(256), cgrd(1024);
  cvt_f32_bf16<<<cgrd, cblk, 0, stream>>>(X,  Xb,  MTOT * HIDDEN);
  cvt_f32_bf16<<<cgrd, cblk, 0, stream>>>(Wq, Wqb, HIDDEN * HIDDEN);
  cvt_f32_bf16<<<cgrd, cblk, 0, stream>>>(Wk, Wkb, HIDDEN * KVDIM);
  cvt_f32_bf16<<<cgrd, cblk, 0, stream>>>(Wv, Wvb, HIDDEN * KVDIM);
  cvt_f32_bf16<<<cgrd, cblk, 0, stream>>>(Wo, Wob, HIDDEN * HIDDEN);

  const dim3 blk(128);
  gemm_bias<false><<<dim3(MTOT / 64, HIDDEN / 64), blk, 0, stream>>>(Xb, Wqb, bq, Qb, HIDDEN);
  gemm_bias<false><<<dim3(MTOT / 64, KVDIM / 64),  blk, 0, stream>>>(Xb, Wkb, bk, Kb, KVDIM);
  gemm_bias<false><<<dim3(MTOT / 64, KVDIM / 64),  blk, 0, stream>>>(Xb, Wvb, bv, Vb, KVDIM);

  gqa_attn<<<dim3(SEQ / 64, BATCH * NHEAD), blk, 0, stream>>>(Qb, Kb, Vb, Ob);

  gemm_bias<true><<<dim3(MTOT / 64, HIDDEN / 64), blk, 0, stream>>>(Ob, Wob, bo, out, HIDDEN);
}
